// ConvGRUAttentionAutoencoder_53541062312025
// MI455X (gfx1250) — matrix operand order repaired and checked by reading, not compiled, not run on hardware
//
#include <hip/hip_runtime.h>
#include <hip/hip_bf16.h>
#include <stdint.h>

// ---------------------------------------------------------------------------
// MI455X / gfx1250 implementation of ConvGRU-Attention autoencoder.
// - All GEMM-shaped work (conv-as-GEMM inside the GRU cells, MHA projections)
//   goes through v_wmma_f32_16x16x32_bf16 (wave32, f32 accumulate).
// - Conv 3-tap is expressed as 3 shifted GEMMs -> branch-free im2col.
// - Dense-GEMM tile staging uses the Tensor Data Mover (tensor_load_to_lds)
//   with D# LDS-padding (16dw data + 8dw pad = 48-element row stride) and
//   OOB-row zero fill, synchronized with s_wait_tensorcnt.
// ---------------------------------------------------------------------------

typedef __bf16 bf16;
typedef __attribute__((ext_vector_type(16))) __bf16   v16bf;
typedef __attribute__((ext_vector_type(8)))  float    v8f;
typedef __attribute__((ext_vector_type(4)))  uint32_t v4u;
typedef __attribute__((ext_vector_type(4)))  int      v4i;
typedef __attribute__((ext_vector_type(8)))  int      v8i;

#define BATCH 64
#define SEQT  128
#define FD    64          // spatial/feature length (conv runs over this dim)
#define EMB   4096        // H2 * FD
#define HEADS 4
#define DH    1024        // EMB / HEADS
#define CMAX  192         // max concat channels (dec0: 64+64+64)

#if __has_builtin(__builtin_amdgcn_tensor_load_to_lds)
#define USE_TDM 1
#else
#define USE_TDM 0
#endif

union FragBF { v16bf v; uint32_t u[8]; };

static __device__ __forceinline__ v8f wmma_bf(v16bf a, v16bf b, v8f c) {
  return __builtin_amdgcn_wmma_f32_16x16x32_bf16(
      /*neg_a=*/false, a, /*neg_b=*/false, b,
      /*c_mod=*/(short)0, c, /*reuse_a=*/false, /*reuse_b=*/false);
}

static __device__ __forceinline__ v8f vzero8() {
  v8f z = {0.f, 0.f, 0.f, 0.f, 0.f, 0.f, 0.f, 0.f};
  return z;
}

// Load A fragment (16x32 bf16, ISA 7.12.2 layout) from a row-major [*][stride]
// matrix: element pairs (2v,2v+1) are K-contiguous -> 8 dword loads.
// dword indices: {0,1,2,3, 8,9,10,11} + (lane>=16 ? 4 : 0)
static __device__ __forceinline__ v16bf load_a_frag(const uint32_t* row,
                                                    int lane) {
  FragBF a;
  const int h4 = (lane & 16) ? 4 : 0;
  a.u[0] = row[h4 + 0];
  a.u[1] = row[h4 + 1];
  a.u[2] = row[h4 + 2];
  a.u[3] = row[h4 + 3];
  a.u[4] = row[h4 + 8];
  a.u[5] = row[h4 + 9];
  a.u[6] = row[h4 + 10];
  a.u[7] = row[h4 + 11];
  return a.v;
}

// Load B fragment (32x16 bf16): lane holds N=lane&15, same K order as A:
// dword indices {0,1,2,3, 8,9,10,11} + (lane>=16 ? 4 : 0).
static __device__ __forceinline__ v16bf load_b_frag(const uint32_t* row,
                                                    int lane) {
  FragBF b;
  const int h4 = (lane & 16) ? 4 : 0;
#pragma unroll
  for (int e = 0; e < 8; ++e) b.u[e] = row[h4 + (e & 3) + ((e & 4) ? 8 : 0)];
  return b.v;
}

// ---- Tensor Data Mover: DMA a [tileRows x 32] bf16 tile from a row-major
// [rows x Kelems] global tensor into LDS, padding each row 32->48 elements
// (pad_interval=3: 16 dwords, pad_amount=7: 8 dwords).  OOB rows read zero.
// D# layout per CDNA5 ISA (group0 + group1, 2D tensor).
static __device__ __forceinline__ void tdm_load_tile(const bf16* gptr,
                                                     void* ldsptr, int Kelems,
                                                     int rows, int tileRows) {
#if USE_TDM
  const uint64_t ga = (uint64_t)(uintptr_t)gptr;
  const uint32_t lds = (uint32_t)(uintptr_t)ldsptr;  // low 32 bits = LDS addr
  v4u g0;
  g0.x = 1u;                                            // count=1 (valid D#)
  g0.y = lds;                                           // lds_addr
  g0.z = (uint32_t)ga;                                  // global_addr[31:0]
  g0.w = (uint32_t)((ga >> 32) & 0x01FFFFFFu) | (2u << 30);  // [56:32]|type=2
  v8i g1;
  g1[0] = (int)((1u << 16) |            // data_size = 1 (2 bytes)
                (1u << 20) |            // pad_enable
                (3u << 22) |            // pad_interval: 16 dwords
                (7u << 25));            // pad_amount:   8 dwords
  g1[1] = (int)(((uint32_t)Kelems & 0xFFFFu) << 16);           // dim0[15:0]
  g1[2] = (int)((((uint32_t)Kelems >> 16) & 0xFFFFu) |         // dim0[31:16]
                (((uint32_t)rows & 0xFFFFu) << 16));           // dim1[15:0]
  g1[3] = (int)((((uint32_t)rows >> 16) & 0xFFFFu) |           // dim1[31:16]
                (32u << 16));                                  // tile_dim0
  g1[4] = tileRows;                                            // tile_dim1
  g1[5] = Kelems;                                              // dim0_stride
  g1[6] = 0;
  g1[7] = 0;
  v4i z4 = {0, 0, 0, 0};
#if defined(__clang_major__) && (__clang_major__ >= 23)
  v8i z8 = {0, 0, 0, 0, 0, 0, 0, 0};
  __builtin_amdgcn_tensor_load_to_lds(g0, g1, z4, z4, z8, 0);
#else
  __builtin_amdgcn_tensor_load_to_lds(g0, g1, z4, z4, 0);
#endif
#else
  (void)gptr; (void)ldsptr; (void)Kelems; (void)rows; (void)tileRows;
#endif
}

// ---- prep kernels ----------------------------------------------------------

__global__ void cast_f32_to_bf16_kernel(const float* __restrict__ src,
                                        bf16* __restrict__ dst, size_t n) {
  size_t i = (size_t)blockIdx.x * 256u + threadIdx.x;
  if (i < n) dst[i] = (bf16)src[i];
}

// Pack conv weight [M][Ctot][3][3] -> [3][M][Cpad] bf16 (tap-major), taking
// kw=1 (W dim is 1 with SAME padding; only center column contributes).
// Channels c >= Ctot are zero-filled so the GEMM needs no K guard.
__global__ void pack_conv_kernel(const float* __restrict__ w,
                                 bf16* __restrict__ dst,
                                 int M, int Ctot, int Cpad) {
  int i = blockIdx.x * 256 + threadIdx.x;
  if (i >= 3 * M * Cpad) return;
  int dk = i / (M * Cpad);
  int rem = i - dk * M * Cpad;
  int m = rem / Cpad, c = rem - m * Cpad;
  float v = 0.f;
  if (c < Ctot) v = w[((m * Ctot + c) * 3 + dk) * 3 + 1];
  dst[i] = (bf16)v;
}

// ---- ConvGRU layer ---------------------------------------------------------
// One block per batch element; 8 wave32s; loops over T internally.
// in0: [B][T][Cin0][FD] bf16  (time-varying input channels)
// aux: [B][Cin1*FD]     bf16  (time-invariant broadcast channels, or null)
// Wg : [3][2*Ch][Cpad] bf16 packed gates weight ; Wc: [3][Ch][Cpad]
// y  : [B][T][Ch][FD] bf16 outputs; optional final-h f32 / bf16 dumps.
__global__ __launch_bounds__(256)
void gru_layer_kernel(const bf16* __restrict__ in0, int Cin0,
                      const bf16* __restrict__ aux, int Cin1,
                      const bf16* __restrict__ Wg, const float* __restrict__ bg,
                      const bf16* __restrict__ Wc, const float* __restrict__ bc,
                      int Ch, int Cpad,
                      const float* __restrict__ h_init,
                      bf16* __restrict__ y,
                      float* __restrict__ h_out_f32,
                      bf16* __restrict__ h_out_bf) {
  // Transposed concat buffer: [spatial + halo][channel].  Row f+1 holds input
  // position f; rows 0 and FD+1 stay zero (conv halo).  Channels >= Ctot stay
  // zero (K padding) -> completely branch-free fragment gathers.
  __shared__ bf16  ldsInT[FD + 2][CMAX];
  __shared__ float hbuf[64][FD];  // recurrent state (f32)
  __shared__ bf16  rbuf[64][FD];  // reset gate
  __shared__ bf16  ubuf[64][FD];  // update gate

  const int b    = blockIdx.x;
  const int tid  = threadIdx.x;
  const int lane = tid & 31;
  const int wave = tid >> 5;
  const int Cin  = Cin0 + Cin1;
  const int KC   = Cpad >> 5;            // #32-wide K chunks per tap
  const int Mg   = 2 * Ch;               // gate rows
  const int nTg  = (Mg >> 4) * 4;        // gates 16x16 tiles (mult of 8)
  const int nTc  = (Ch >> 4) * 4;        // cand  16x16 tiles (mult of 8)

  // zero whole concat buffer once (halo rows + padded channels stay zero)
  for (int i = tid; i < (FD + 2) * CMAX; i += 256)
    ((bf16*)ldsInT)[i] = (bf16)0.f;
  // init recurrent state
  for (int i = tid; i < Ch * FD; i += 256)
    hbuf[i / FD][i % FD] = h_init ? h_init[(size_t)b * Ch * FD + i] : 0.f;
  __syncthreads();
  // time-invariant aux channels staged once
  if (aux)
    for (int i = tid; i < Cin1 * FD; i += 256)
      ldsInT[1 + (i % FD)][Cin0 + i / FD] = aux[(size_t)b * Cin1 * FD + i];
  __syncthreads();

  for (int t = 0; t < SEQT; ++t) {
    // stage x_t and h (as bf16) into the transposed concat buffer
    for (int i = tid; i < Cin0 * FD; i += 256)
      ldsInT[1 + (i % FD)][i / FD] =
          in0[((size_t)b * SEQT + t) * Cin0 * FD + i];
    for (int i = tid; i < Ch * FD; i += 256)
      ldsInT[1 + (i % FD)][Cin + i / FD] = (bf16)hbuf[i / FD][i % FD];
    __syncthreads();

    // ---- gates GEMM: 3 shifted taps, K = channels ------------------------
    for (int tile = wave; tile < nTg; tile += 8) {
      int tm = tile >> 2, tn = tile & 3;
      int m = tm * 16 + (lane & 15);
      int n = tn * 16 + (lane & 15);
      v8f acc = vzero8();
#pragma unroll
      for (int dk = 0; dk < 3; ++dk) {
        for (int kc = 0; kc < KC; ++kc) {
          v16bf af = load_a_frag(
              (const uint32_t*)&Wg[((size_t)(dk * Mg + m)) * Cpad + kc * 32],
              lane);
          v16bf bf_ = load_b_frag(
              (const uint32_t*)&ldsInT[n + dk][kc * 32], lane);
          acc = wmma_bf(af, bf_, acc);
        }
      }
      int nn = tn * 16 + (lane & 15);
#pragma unroll
      for (int e = 0; e < 8; ++e) {
        int mm = tm * 16 + e + ((lane & 16) ? 8 : 0);
        float g = acc[e] + bg[mm];
        float s = 1.f / (1.f + __expf(-g));
        if (mm < Ch) rbuf[mm][nn] = (bf16)s;
        else         ubuf[mm - Ch][nn] = (bf16)s;
      }
    }
    __syncthreads();

    // replace h-channels with r*h for the candidate conv
    for (int i = tid; i < Ch * FD; i += 256) {
      int c = i / FD, f = i % FD;
      ldsInT[1 + f][Cin + c] = (bf16)((float)rbuf[c][f] * hbuf[c][f]);
    }
    __syncthreads();

    // ---- candidate GEMM + blend: h' = (1-u)h + u*tanh(...) ---------------
    for (int tile = wave; tile < nTc; tile += 8) {
      int tm = tile >> 2, tn = tile & 3;
      int m = tm * 16 + (lane & 15);
      int n = tn * 16 + (lane & 15);
      v8f acc = vzero8();
#pragma unroll
      for (int dk = 0; dk < 3; ++dk) {
        for (int kc = 0; kc < KC; ++kc) {
          v16bf af = load_a_frag(
              (const uint32_t*)&Wc[((size_t)(dk * Ch + m)) * Cpad + kc * 32],
              lane);
          v16bf bf_ = load_b_frag(
              (const uint32_t*)&ldsInT[n + dk][kc * 32], lane);
          acc = wmma_bf(af, bf_, acc);
        }
      }
      int nn = tn * 16 + (lane & 15);
#pragma unroll
      for (int e = 0; e < 8; ++e) {
        int mm = tm * 16 + e + ((lane & 16) ? 8 : 0);
        float cand = tanhf(acc[e] + bc[mm]);
        float u = (float)ubuf[mm][nn];
        float hn = (1.f - u) * hbuf[mm][nn] + u * cand;
        hbuf[mm][nn] = hn;  // unique (m,n) owner: no race
        y[(((size_t)b * SEQT + t) * Ch + mm) * FD + nn] = (bf16)hn;
      }
    }
    __syncthreads();
  }

  for (int i = tid; i < Ch * FD; i += 256) {
    float hv = hbuf[i / FD][i % FD];
    if (h_out_f32) h_out_f32[(size_t)b * Ch * FD + i] = hv;
    if (h_out_bf)  h_out_bf[(size_t)b * Ch * FD + i] = (bf16)hv;
  }
}

// ---- dense GEMM: C[M][N] = A[M][K] * W[N][K]^T + bias ----------------------
// block tile 128x64, 8 waves of 32x32 (2x2 WMMA tiles).  Tiles are staged
// into LDS by the Tensor Data Mover (wave 0 issues, s_wait_tensorcnt syncs);
// the D# pad feature produces the 48-element row stride, OOB rows read zero.
__global__ __launch_bounds__(256)
void gemm_bf16_kernel(const bf16* __restrict__ A, const bf16* __restrict__ W,
                      const float* __restrict__ bias, bf16* __restrict__ C,
                      int M, int N, int K) {
  __shared__ bf16 Atile[128][48];  // 96B row stride: 16B-aligned rows
  __shared__ bf16 Wtile[64][48];
  const int tid = threadIdx.x, lane = tid & 31, wave = tid >> 5;
  const int wm = wave >> 1, wn = wave & 1;
  const int m0 = blockIdx.x * 128, n0 = blockIdx.y * 64;

  v8f acc[2][2];
#pragma unroll
  for (int i = 0; i < 2; ++i)
#pragma unroll
    for (int j = 0; j < 2; ++j) acc[i][j] = vzero8();

  for (int kc = 0; kc < K; kc += 32) {
#if USE_TDM
    if (wave == 0) {
      // async DMA both tiles; OOB rows (m >= M) are zero-filled by TDM
      tdm_load_tile(A + (size_t)m0 * K + kc, &Atile[0][0], K, M, 128);
      tdm_load_tile(W + (size_t)n0 * K + kc, &Wtile[0][0], K, N, 64);
      __builtin_amdgcn_s_wait_tensorcnt(0);
    }
    __syncthreads();
#else
    const uint4 zero4 = {0u, 0u, 0u, 0u};
    // stage A (128x32) as 512 x b128: 2 per thread
#pragma unroll
    for (int s = 0; s < 2; ++s) {
      int q = tid + s * 256;
      int r = q >> 2, cq = (q & 3) * 8;
      int gm = m0 + r;
      *(uint4*)&Atile[r][cq] =
          (gm < M) ? *(const uint4*)&A[(size_t)gm * K + kc + cq] : zero4;
    }
    // stage W (64x32) as 256 x b128: 1 per thread
    {
      int r = tid >> 2, cq = (tid & 3) * 8;
      *(uint4*)&Wtile[r][cq] =
          *(const uint4*)&W[(size_t)(n0 + r) * K + kc + cq];
    }
    __syncthreads();
#endif
    if (kc + 32 < K) {  // hint next K slab into the caches (global_prefetch_b8)
      int pm = m0 + (tid >> 1);
      if (pm >= M) pm = M - 1;
      __builtin_prefetch(A + (size_t)pm * K + kc + 32, 0, 1);
      __builtin_prefetch(W + (size_t)(n0 + (tid & 63)) * K + kc + 32, 0, 1);
    }
#pragma unroll
    for (int i = 0; i < 2; ++i) {
      int m = wm * 32 + i * 16 + (lane & 15);
      v16bf af = load_a_frag((const uint32_t*)&Atile[m][0], lane);
#pragma unroll
      for (int j = 0; j < 2; ++j) {
        int n = wn * 32 + j * 16 + (lane & 15);
        v16bf bfr = load_b_frag((const uint32_t*)&Wtile[n][0], lane);
        acc[i][j] = wmma_bf(af, bfr, acc[i][j]);
      }
    }
    __syncthreads();
  }

#pragma unroll
  for (int i = 0; i < 2; ++i)
#pragma unroll
    for (int j = 0; j < 2; ++j)
#pragma unroll
      for (int e = 0; e < 8; ++e) {
        int m = m0 + wm * 32 + i * 16 + e + ((lane & 16) ? 8 : 0);
        int n = n0 + wn * 32 + j * 16 + (lane & 15);
        if (m < M) C[(size_t)m * N + n] = (bf16)(acc[i][j][e] + bias[n]);
      }
}

// ---- attention core: scores -> softmax -> context (per batch*head) --------
__global__ __launch_bounds__(256)
void attn_softmax_ctx_kernel(const bf16* __restrict__ q,
                             const bf16* __restrict__ Kp,
                             const bf16* __restrict__ Vp,
                             bf16* __restrict__ ctx) {
  __shared__ float sc[SEQT];
  __shared__ float stat;
  const int b = blockIdx.x >> 2, h = blockIdx.x & 3;
  const int tid = threadIdx.x;
  const bf16* qh = q + (size_t)b * EMB + h * DH;
  if (tid < SEQT) {
    const bf16* kh = Kp + ((size_t)b * SEQT + tid) * EMB + h * DH;
    float s = 0.f;
    for (int d = 0; d < DH; ++d) s += (float)qh[d] * (float)kh[d];
    sc[tid] = s * 0.03125f;  // 1/sqrt(1024)
  }
  __syncthreads();
  if (tid == 0) {
    float mx = sc[0];
    for (int t = 1; t < SEQT; ++t) mx = fmaxf(mx, sc[t]);
    float sum = 0.f;
    for (int t = 0; t < SEQT; ++t) {
      float e = __expf(sc[t] - mx);
      sc[t] = e;
      sum += e;
    }
    stat = 1.f / sum;
  }
  __syncthreads();
  float inv = stat;
  for (int d = tid; d < DH; d += 256) {
    float acc = 0.f;
    for (int t = 0; t < SEQT; ++t)
      acc += sc[t] * (float)Vp[((size_t)b * SEQT + t) * EMB + h * DH + d];
    ctx[(size_t)b * EMB + h * DH + d] = (bf16)(acc * inv);
  }
}

// ---- final 1x1 conv (channel contraction) ----------------------------------
__global__ void final_proj_kernel(const bf16* __restrict__ z1,
                                  const float* __restrict__ wfin,
                                  const float* __restrict__ bfin,
                                  float* __restrict__ out) {
  int idx = blockIdx.x * 256 + threadIdx.x;  // b*T*F flat
  if (idx >= BATCH * SEQT * FD) return;
  int f = idx & 63, bt = idx >> 6;
  float s = bfin[0];
  const bf16* p = z1 + (size_t)bt * 32 * FD + f;
#pragma unroll
  for (int c = 0; c < 32; ++c) s += wfin[c] * (float)p[c * FD];
  out[idx] = s;
}

// ---------------------------------------------------------------------------
extern "C" void kernel_launch(void* const* d_in, const int* in_sizes, int n_in,
                              void* d_out, int out_size, void* d_ws,
                              size_t ws_size, hipStream_t stream) {
  (void)in_sizes; (void)n_in; (void)out_size; (void)ws_size;
  const float* x     = (const float*)d_in[0];
  const float* wg_e0 = (const float*)d_in[1];
  const float* bg_e0 = (const float*)d_in[2];
  const float* wc_e0 = (const float*)d_in[3];
  const float* bc_e0 = (const float*)d_in[4];
  const float* wg_e1 = (const float*)d_in[5];
  const float* bg_e1 = (const float*)d_in[6];
  const float* wc_e1 = (const float*)d_in[7];
  const float* bc_e1 = (const float*)d_in[8];
  const float* wg_d0 = (const float*)d_in[9];
  const float* bg_d0 = (const float*)d_in[10];
  const float* wc_d0 = (const float*)d_in[11];
  const float* bc_d0 = (const float*)d_in[12];
  const float* wg_d1 = (const float*)d_in[13];
  const float* bg_d1 = (const float*)d_in[14];
  const float* wc_d1 = (const float*)d_in[15];
  const float* bc_d1 = (const float*)d_in[16];
  const float* w_qkv = (const float*)d_in[17];
  const float* b_qkv = (const float*)d_in[18];
  const float* w_out = (const float*)d_in[19];
  const float* b_out = (const float*)d_in[20];
  const float* w_fin = (const float*)d_in[21];
  const float* b_fin = (const float*)d_in[22];
  float* out = (float*)d_out;

  uint8_t* ws = (uint8_t*)d_ws;
  size_t off = 0;
  auto alloc = [&](size_t bytes) -> void* {
    void* p = ws + off;
    off = (off + bytes + 255) & ~(size_t)255;
    return p;
  };

  bf16* xbf    = (bf16*)alloc((size_t)BATCH * SEQT * FD * 2);
  bf16* wqkvbf = (bf16*)alloc((size_t)3 * EMB * EMB * 2);
  bf16* woutbf = (bf16*)alloc((size_t)EMB * EMB * 2);
  bf16* wge0p  = (bf16*)alloc((size_t)3 * 64 * 64 * 2);
  bf16* wce0p  = (bf16*)alloc((size_t)3 * 32 * 64 * 2);
  bf16* wge1p  = (bf16*)alloc((size_t)3 * 128 * 96 * 2);
  bf16* wce1p  = (bf16*)alloc((size_t)3 * 64 * 96 * 2);
  bf16* wgd0p  = (bf16*)alloc((size_t)3 * 128 * 192 * 2);
  bf16* wcd0p  = (bf16*)alloc((size_t)3 * 64 * 192 * 2);
  bf16* wgd1p  = (bf16*)alloc((size_t)3 * 64 * 96 * 2);
  bf16* wcd1p  = (bf16*)alloc((size_t)3 * 32 * 96 * 2);
  bf16* y0     = (bf16*)alloc((size_t)BATCH * SEQT * 32 * FD * 2);
  bf16* y1     = (bf16*)alloc((size_t)BATCH * SEQT * 64 * FD * 2);
  float* h0f   = (float*)alloc((size_t)BATCH * 32 * FD * 4);
  float* h1f   = (float*)alloc((size_t)BATCH * 64 * FD * 4);
  bf16* h1bf   = (bf16*)alloc((size_t)BATCH * EMB * 2);
  bf16* qproj  = (bf16*)alloc((size_t)BATCH * EMB * 2);
  bf16* Kp     = (bf16*)alloc((size_t)BATCH * SEQT * EMB * 2);
  bf16* Vp     = (bf16*)alloc((size_t)BATCH * SEQT * EMB * 2);
  bf16* ctxb   = (bf16*)alloc((size_t)BATCH * EMB * 2);
  bf16* ctxo   = (bf16*)alloc((size_t)BATCH * EMB * 2);
  bf16* z0     = (bf16*)alloc((size_t)BATCH * SEQT * 64 * FD * 2);
  bf16* z1     = (bf16*)alloc((size_t)BATCH * SEQT * 32 * FD * 2);

  auto cast = [&](const float* s, bf16* d, size_t n) {
    cast_f32_to_bf16_kernel<<<(unsigned)((n + 255) / 256), 256, 0, stream>>>(
        s, d, n);
  };
  auto pack = [&](const float* w, bf16* d, int M, int Ct, int Cp) {
    pack_conv_kernel<<<(3 * M * Cp + 255) / 256, 256, 0, stream>>>(w, d, M, Ct,
                                                                   Cp);
  };

  // ---- prep ----------------------------------------------------------------
  cast(x, xbf, (size_t)BATCH * SEQT * FD);
  cast(w_qkv, wqkvbf, (size_t)3 * EMB * EMB);
  cast(w_out, woutbf, (size_t)EMB * EMB);
  pack(wg_e0, wge0p, 64, 33, 64);
  pack(wc_e0, wce0p, 32, 33, 64);
  pack(wg_e1, wge1p, 128, 96, 96);
  pack(wc_e1, wce1p, 64, 96, 96);
  pack(wg_d0, wgd0p, 128, 192, 192);
  pack(wc_d0, wcd0p, 64, 192, 192);
  pack(wg_d1, wgd1p, 64, 96, 96);
  pack(wc_d1, wcd1p, 32, 96, 96);

  // ---- encoder -------------------------------------------------------------
  gru_layer_kernel<<<BATCH, 256, 0, stream>>>(xbf, 1, nullptr, 0,
      wge0p, bg_e0, wce0p, bc_e0, 32, 64, nullptr, y0, h0f, nullptr);
  gru_layer_kernel<<<BATCH, 256, 0, stream>>>(y0, 32, nullptr, 0,
      wge1p, bg_e1, wce1p, bc_e1, 64, 96, nullptr, y1, h1f, h1bf);

  // ---- attention bottleneck ------------------------------------------------
  const bf16* wq = wqkvbf;
  const bf16* wk = wqkvbf + (size_t)EMB * EMB;
  const bf16* wv = wqkvbf + (size_t)2 * EMB * EMB;
  gemm_bf16_kernel<<<dim3(1, EMB / 64), 256, 0, stream>>>(
      h1bf, wq, b_qkv, qproj, BATCH, EMB, EMB);
  gemm_bf16_kernel<<<dim3(BATCH * SEQT / 128, EMB / 64), 256, 0, stream>>>(
      y1, wk, b_qkv + EMB, Kp, BATCH * SEQT, EMB, EMB);
  gemm_bf16_kernel<<<dim3(BATCH * SEQT / 128, EMB / 64), 256, 0, stream>>>(
      y1, wv, b_qkv + 2 * EMB, Vp, BATCH * SEQT, EMB, EMB);
  attn_softmax_ctx_kernel<<<BATCH * HEADS, 256, 0, stream>>>(qproj, Kp, Vp,
                                                             ctxb);
  gemm_bf16_kernel<<<dim3(1, EMB / 64), 256, 0, stream>>>(
      ctxb, woutbf, b_out, ctxo, BATCH, EMB, EMB);

  // ---- decoder (input = concat[y1, broadcast ctx]; h init = enc states) ---
  gru_layer_kernel<<<BATCH, 256, 0, stream>>>(y1, 64, ctxo, 64,
      wgd0p, bg_d0, wcd0p, bc_d0, 64, 192, h1f, z0, nullptr, nullptr);
  gru_layer_kernel<<<BATCH, 256, 0, stream>>>(z0, 64, nullptr, 0,
      wgd1p, bg_d1, wcd1p, bc_d1, 32, 96, h0f, z1, nullptr, nullptr);

  // ---- final 1x1 conv ------------------------------------------------------
  final_proj_kernel<<<(BATCH * SEQT * FD + 255) / 256, 256, 0, stream>>>(
      z1, w_fin, b_fin, out);
}
